// GroupedQueryAttention_32409823216252
// MI455X (gfx1250) — 16-bit and 8-bit matrix operand order repaired and checked by reading, compiled, not run on hardware
//
#include <hip/hip_runtime.h>
#include <hip/hip_bf16.h>
#include <stdint.h>

#define B_   2
#define S_   2048
#define D_   2048
#define H_   16
#define KV_  4
#define DH_  128
#define G_   (H_ / KV_)
#define TOK  (B_ * S_)
#define DKV  (KV_ * DH_)

typedef int      v8i  __attribute__((ext_vector_type(8)));
typedef float    v8f  __attribute__((ext_vector_type(8)));
typedef _Float16 v16h __attribute__((ext_vector_type(16)));

union AFrag { v16h v; unsigned int u[8]; };

static __device__ __forceinline__ int imin(int a, int b) { return a < b ? a : b; }

__global__ void abs_partial_k(const float* __restrict__ W, int n, float* __restrict__ partial) {
    __shared__ float red[256];
    float s = 0.f;
    for (int i = blockIdx.x * blockDim.x + threadIdx.x; i < n; i += gridDim.x * blockDim.x)
        s += fabsf(W[i]);
    red[threadIdx.x] = s;
    __syncthreads();
    for (int off = 128; off > 0; off >>= 1) {
        if ((int)threadIdx.x < off) red[threadIdx.x] += red[threadIdx.x + off];
        __syncthreads();
    }
    if (threadIdx.x == 0) partial[blockIdx.x] = red[0];
}

__global__ void finalize_sw_k(const float* __restrict__ partial, int np, float invn,
                              float* __restrict__ swout) {
    if (threadIdx.x == 0) {
        float s = 0.f;
        for (int i = 0; i < np; ++i) s += partial[i];
        *swout = s * invn + 1e-5f;
    }
}

__global__ void quant_w_k(const float* __restrict__ W, signed char* __restrict__ Wt,
                          int n, const float* __restrict__ swp) {
    float inv = 1.f / (*swp);
    for (int i = blockIdx.x * blockDim.x + threadIdx.x; i < n; i += gridDim.x * blockDim.x) {
        float r = rintf(W[i] * inv);
        r = fminf(1.f, fmaxf(-1.f, r));
        Wt[i] = (signed char)r;
    }
}

__global__ void quant_x_k(const float* __restrict__ x, signed char* __restrict__ xq,
                          float* __restrict__ sx, int K) {
    __shared__ float red[256];
    const float* row = x + (size_t)blockIdx.x * K;
    float mx = 0.f;
    for (int i = threadIdx.x; i < K; i += blockDim.x) mx = fmaxf(mx, fabsf(row[i]));
    red[threadIdx.x] = mx;
    __syncthreads();
    for (int off = 128; off > 0; off >>= 1) {
        if ((int)threadIdx.x < off) red[threadIdx.x] = fmaxf(red[threadIdx.x], red[threadIdx.x + off]);
        __syncthreads();
    }
    float s = 127.f / fmaxf(red[0], 1e-5f);
    if (threadIdx.x == 0) sx[blockIdx.x] = s;
    signed char* orow = xq + (size_t)blockIdx.x * K;
    for (int i = threadIdx.x; i < K; i += blockDim.x) {
        float r = rintf(row[i] * s);
        r = fminf(127.f, fmaxf(-128.f, r));
        orow[i] = (signed char)r;
    }
}

__global__ void __launch_bounds__(256)
gemm_iu8_k(const signed char* __restrict__ A, const signed char* __restrict__ W,
           const float* __restrict__ sx, const float* __restrict__ swp,
           float* __restrict__ C, int M, int N, int K) {
    int wid  = threadIdx.x >> 5;
    int lane = threadIdx.x & 31;
    int hi = lane >> 4, ln = lane & 15;
    int nt64 = N >> 6;
    int tile = blockIdx.x * (blockDim.x >> 5) + wid;
    int tm = tile / nt64, tn = tile % nt64;
    if (tm >= (M >> 4)) return;

    const signed char* arow = A + (size_t)(tm * 16 + ln) * K;
    const signed char* brow = W + (size_t)(tn * 64 + ln) * K;

    v8i acc[4] = {};
    for (int kb = 0; kb < K; kb += 64) {
        v8i a;
#pragma unroll
        for (int j = 0; j < 8; ++j) {
            int ka = kb + ((j >> 1) << 4) + ((j & 1) << 2) + (hi << 3);
            a[j] = *(const int*)(arow + ka);
        }
#pragma unroll
        for (int t = 0; t < 4; ++t) {
            v8i b;
#pragma unroll
            for (int j = 0; j < 8; ++j) {
                int kw = kb + ((j >> 1) << 4) + ((j & 1) << 2) + (hi << 3);
                b[j] = *(const int*)(brow + (size_t)(t * 16) * K + kw);
            }
            acc[t] = __builtin_amdgcn_wmma_i32_16x16x64_iu8(true, a, true, b, acc[t], false, false);
        }
    }

    float sw = *swp;
#pragma unroll
    for (int r = 0; r < 8; ++r) {
        int m = tm * 16 + r + hi * 8;
        float dq = sw / sx[m];
#pragma unroll
        for (int t = 0; t < 4; ++t)
            C[(size_t)m * N + tn * 64 + t * 16 + ln] = (float)acc[t][r] * dq;
    }
}

__global__ void rope_qk_k(const float* __restrict__ inp, _Float16* __restrict__ outp, int nheads) {
    int idx = blockIdx.x * blockDim.x + threadIdx.x;
    int total = B_ * nheads * S_ * (DH_ / 2);
    if (idx >= total) return;
    int i = idx % (DH_ / 2); int t = idx / (DH_ / 2);
    int s = t % S_; t /= S_;
    int h = t % nheads; int b = t / nheads;
    float freq = __expf(-(float)(2 * i) * (9.210340371976184f / (float)DH_));
    float ang = (float)s * freq;
    float c = __cosf(ang), sn = __sinf(ang);
    const float* row = inp + ((size_t)(b * S_ + s) * nheads + h) * DH_;
    float x1 = row[i], x2 = row[i + DH_ / 2];
    _Float16* orow = outp + ((size_t)(b * nheads + h) * S_ + s) * DH_;
    orow[i]            = (_Float16)(x1 * c - x2 * sn);
    orow[i + DH_ / 2]  = (_Float16)(x2 * c + x1 * sn);
}

__global__ void v_tr_k(const float* __restrict__ vf, _Float16* __restrict__ vt) {
    int idx = blockIdx.x * blockDim.x + threadIdx.x;
    int total = B_ * KV_ * S_ * DH_;
    if (idx >= total) return;
    int d = idx % DH_; int t = idx / DH_;
    int s = t % S_; t /= S_;
    int kv = t % KV_; int b = t / KV_;
    float val = vf[((size_t)(b * S_ + s) * KV_ + kv) * DH_ + d];
    vt[((size_t)(b * KV_ + kv) * DH_ + d) * S_ + s] = (_Float16)val;
}

__global__ void __launch_bounds__(128)
attn_k(const _Float16* __restrict__ qh, const _Float16* __restrict__ kh,
       const _Float16* __restrict__ vt, float* __restrict__ outp) {
    __shared__ float    sc_all[4][16][32];
    __shared__ _Float16 pb_all[4][16][32];
    __shared__ float    rs_all[4][16];

    int wid = threadIdx.x >> 5;
    float    (*sc)[32] = sc_all[wid];
    _Float16 (*pb)[32] = pb_all[wid];
    float*   rs        = rs_all[wid];

    int id = blockIdx.x * 4 + wid;
    int qb = id % (S_ / 16); id /= (S_ / 16);
    int h  = id % H_;        int b = id / H_;
    int kv = h / G_;
    int lane = threadIdx.x & 31;
    int hi = lane >> 4, ln = lane & 15;

    const _Float16* qbase = qh + ((size_t)(b * H_ + h) * S_ + qb * 16) * DH_;
    const _Float16* kbase = kh + (size_t)(b * KV_ + kv) * S_ * DH_;
    const _Float16* vbase = vt + (size_t)(b * KV_ + kv) * DH_ * S_;

    AFrag qa[4];
#pragma unroll
    for (int c = 0; c < 4; ++c)
#pragma unroll
        for (int j = 0; j < 8; ++j) {
            int k = c * 32 + ((j >> 2) << 4) + ((j & 3) << 1) + hi * 8;
            qa[c].u[j] = *(const unsigned int*)(qbase + (size_t)ln * DH_ + k);
        }

    v8f o[8] = {};
    float m = -1e30f, l = 0.f;
    const float scale = 0.08838834764831843f;
    int qend = qb * 16 + 16;

    for (int kb = 0; kb < qend; kb += 32) {
        bool second = (kb + 16) < qend;
        __builtin_prefetch(kbase + (size_t)imin(kb + 32 + ln, S_ - 1) * DH_, 0, 1);

        v8f s0 = {}, s1 = {};
#pragma unroll
        for (int c = 0; c < 4; ++c) {
            AFrag b0, b1;
#pragma unroll
            for (int j = 0; j < 8; ++j) {
                int kk = c * 32 + ((j >> 2) << 4) + ((j & 3) << 1) + hi * 8;
                int t0 = kb + ln;
                int t1 = imin(kb + 16 + ln, S_ - 1);
                b0.u[j] = *(const unsigned int*)(kbase + (size_t)t0 * DH_ + kk);
                b1.u[j] = *(const unsigned int*)(kbase + (size_t)t1 * DH_ + kk);
            }
            s0 = __builtin_amdgcn_wmma_f32_16x16x32_f16(false, qa[c].v, false, b0.v,
                                                        (short)0, s0, false, false);
            if (second)
                s1 = __builtin_amdgcn_wmma_f32_16x16x32_f16(false, qa[c].v, false, b1.v,
                                                            (short)0, s1, false, false);
        }

#pragma unroll
        for (int r = 0; r < 8; ++r) {
            int row_t = qb * 16 + r + hi * 8;
            int c0 = kb + ln, c1 = kb + 16 + ln;
            sc[r + hi * 8][ln]      = (c0 <= row_t) ? s0[r] * scale : -1e30f;
            sc[r + hi * 8][16 + ln] = (second && c1 <= row_t) ? s1[r] * scale : -1e30f;
        }
        asm volatile("s_wait_dscnt 0x0" ::: "memory");

        float cm = -1e30f;
#pragma unroll
        for (int c2 = 0; c2 < 32; ++c2) cm = fmaxf(cm, sc[ln][c2]);
        float mnew = fmaxf(m, cm);
        float resc = __expf(m - mnew);
        float ls = 0.f;
#pragma unroll
        for (int c2 = 0; c2 < 32; ++c2) {
            float p = __expf(sc[ln][c2] - mnew);
            pb[ln][c2] = (_Float16)p;
            ls += p;
        }
        l = l * resc + ls;
        m = mnew;
        rs[ln] = resc;
        asm volatile("s_wait_dscnt 0x0" ::: "memory");

#pragma unroll
        for (int r = 0; r < 8; ++r) {
            float f = rs[r + hi * 8];
#pragma unroll
            for (int t = 0; t < 8; ++t) o[t][r] *= f;
        }

        AFrag pa;
#pragma unroll
        for (int j = 0; j < 8; ++j) {
            int k = ((j >> 2) << 4) + ((j & 3) << 1) + hi * 8;
            pa.u[j] = *(const unsigned int*)&pb[ln][k];
        }
#pragma unroll
        for (int t = 0; t < 8; ++t) {
            AFrag vb;
#pragma unroll
            for (int j = 0; j < 8; ++j) {
                int key = imin(kb + ((j >> 2) << 4) + ((j & 3) << 1) + hi * 8, S_ - 2);
                int d = t * 16 + ln;
                vb.u[j] = *(const unsigned int*)(vbase + (size_t)d * S_ + key);
            }
            o[t] = __builtin_amdgcn_wmma_f32_16x16x32_f16(false, pa.v, false, vb.v,
                                                          (short)0, o[t], false, false);
        }
    }

    rs[ln] = 1.f / l;
    asm volatile("s_wait_dscnt 0x0" ::: "memory");
#pragma unroll
    for (int r = 0; r < 8; ++r) {
        float inv = rs[r + hi * 8];
        int s = qb * 16 + r + hi * 8;
        float* orow = outp + (size_t)(b * S_ + s) * D_ + h * DH_;
#pragma unroll
        for (int t = 0; t < 8; ++t)
            orow[t * 16 + ln] = o[t][r] * inv;
    }
}

extern "C" void kernel_launch(void* const* d_in, const int* in_sizes, int n_in,
                              void* d_out, int out_size, void* d_ws, size_t ws_size,
                              hipStream_t stream) {
    const float* x  = (const float*)d_in[0];
    const float* Wq = (const float*)d_in[1];
    const float* Wk = (const float*)d_in[2];
    const float* Wv = (const float*)d_in[3];
    const float* Wo = (const float*)d_in[4];

    char* ws = (char*)d_ws;
    size_t off = 0;
    auto carve = [&](size_t bytes) -> char* {
        char* p = ws + off;
        off += (bytes + 255) & ~(size_t)255;
        return p;
    };

    signed char* Wqt = (signed char*)carve((size_t)D_ * D_);
    signed char* Wkt = (signed char*)carve((size_t)DKV * D_);
    signed char* Wvt = (signed char*)carve((size_t)DKV * D_);
    signed char* Wot = (signed char*)carve((size_t)D_ * D_);
    float* partials  = (float*)carve(4 * 256 * sizeof(float));
    float* swv       = (float*)carve(4 * sizeof(float));
    signed char* xq  = (signed char*)carve((size_t)TOK * D_);
    float* sx        = (float*)carve((size_t)TOK * sizeof(float));
    float* sx2       = (float*)carve((size_t)TOK * sizeof(float));
    float* qf        = (float*)carve((size_t)TOK * D_ * sizeof(float));
    float* kf        = (float*)carve((size_t)TOK * DKV * sizeof(float));
    float* vf        = (float*)carve((size_t)TOK * DKV * sizeof(float));
    _Float16* qhb    = (_Float16*)carve((size_t)B_ * H_ * S_ * DH_ * 2);
    _Float16* khb    = (_Float16*)carve((size_t)B_ * KV_ * S_ * DH_ * 2);
    _Float16* vtb    = (_Float16*)carve((size_t)B_ * KV_ * S_ * DH_ * 2);
    (void)ws_size;

    const float* Wsrc[4]  = {Wq, Wk, Wv, Wo};
    signed char* Wdst[4]  = {Wqt, Wkt, Wvt, Wot};
    const int    Wn[4]    = {D_ * D_, DKV * D_, DKV * D_, D_ * D_};
    for (int i = 0; i < 4; ++i) {
        abs_partial_k<<<256, 256, 0, stream>>>(Wsrc[i], Wn[i], partials + i * 256);
        finalize_sw_k<<<1, 32, 0, stream>>>(partials + i * 256, 256, 1.f / (float)Wn[i], swv + i);
        quant_w_k<<<256, 256, 0, stream>>>(Wsrc[i], Wdst[i], Wn[i], swv + i);
    }

    quant_x_k<<<TOK, 256, 0, stream>>>(x, xq, sx, D_);

    gemm_iu8_k<<<(TOK / 16) * (D_ / 64) / 8, 256, 0, stream>>>(xq, Wqt, sx, swv + 0, qf, TOK, D_, D_);
    gemm_iu8_k<<<(TOK / 16) * (DKV / 64) / 8, 256, 0, stream>>>(xq, Wkt, sx, swv + 1, kf, TOK, DKV, D_);
    gemm_iu8_k<<<(TOK / 16) * (DKV / 64) / 8, 256, 0, stream>>>(xq, Wvt, sx, swv + 2, vf, TOK, DKV, D_);

    rope_qk_k<<<(B_ * H_ * S_ * (DH_ / 2) + 255) / 256, 256, 0, stream>>>(qf, qhb, H_);
    rope_qk_k<<<(B_ * KV_ * S_ * (DH_ / 2) + 255) / 256, 256, 0, stream>>>(kf, khb, KV_);
    v_tr_k<<<(B_ * KV_ * S_ * DH_ + 255) / 256, 256, 0, stream>>>(vf, vtb);

    float* attn = qf;
    attn_k<<<B_ * H_ * (S_ / 16) / 4, 128, 0, stream>>>(qhb, khb, vtb, attn);

    quant_x_k<<<TOK, 256, 0, stream>>>(attn, xq, sx2, D_);
    gemm_iu8_k<<<(TOK / 16) * (D_ / 64) / 8, 256, 0, stream>>>(xq, Wot, sx2, swv + 3,
                                                               (float*)d_out, TOK, D_, D_);
}
